// Long_LSTM_Top_81535659147365
// MI455X (gfx1250) — compile-verified, matrix operand order checked by reading, not run on hardware
//
#include <hip/hip_runtime.h>
#include <hip/hip_bf16.h>

typedef __attribute__((ext_vector_type(16))) _Float16 half16;
typedef __attribute__((ext_vector_type(8)))  _Float16 half8;
typedef __attribute__((ext_vector_type(8)))  float    float8;

#define B_      256
#define T_      300
#define C_      128
#define H_      256
#define CLS_    60
#define L_      299   // T-1 steps
#define STRIDE_ 145
#define WIN_    154
#define NSEQ_   512   // 2 windows * B
#define KTOT_   384   // C + H fused K dimension
#define G4H_    1024  // 4*H gate columns
#define NBLK_   64    // persistent grid: 4 x 16 blocks
#define BPAD_   392   // LDS B-tile K stride (384 + 8 halves) -> no bank conflicts

// ---------------- workspace layout (bytes, 256-aligned) ----------------
// Wc   f16 [1024][384]        @ 0          (786432)
// bsum f32 [1024]             @ 786432     (4096)
// d    f16 [256][299][128]    @ 790528     (19595264)
// h0   f16 [512][256]         @ 20385792   (262144)
// h1   f16 [512][256]         @ 20647936   (262144)
// hsum f32 [512][256]         @ 20910080   (524288)
// sync u32                    @ 21434368   (256)
// total ~21.4 MB

__device__ __forceinline__ float sigmoidf_(float x) {
    return 1.0f / (1.0f + __expf(-x));
}

// Load a 16x32 / 32x16 f16 WMMA fragment for this lane.
// Lane layout: row = lane%16 (baked into p); K groups [kb,kb+8) and
// [kb+16,kb+24), kb = 8*(lane/16); p points at K=kb. Works for global or
// LDS pointers (addrspace inferred after inlining -> b128 / ds_load_b128).
__device__ __forceinline__ half16 load_frag16(const _Float16* p) {
    half8 lo = *(const half8*)(p);
    half8 hi = *(const half8*)(p + 16);
    half16 r;
#pragma unroll
    for (int i = 0; i < 8; ++i) { r[i] = lo[i]; r[i + 8] = hi[i]; }
    return r;
}

#define WMMA_F16(A, Bf, Cacc) \
    __builtin_amdgcn_wmma_f32_16x16x32_f16(false, (A), false, (Bf), (short)0, (Cacc), false, false)

// ---------------- prep kernels ----------------
__global__ void k_prep_w(const float* __restrict__ W_ih,
                         const float* __restrict__ W_hh,
                         _Float16* __restrict__ Wc) {
    int idx = blockIdx.x * blockDim.x + threadIdx.x;
    if (idx >= G4H_ * KTOT_) return;
    int g = idx / KTOT_, k = idx % KTOT_;
    float v = (k < C_) ? W_ih[g * C_ + k] : W_hh[g * H_ + (k - C_)];
    Wc[idx] = (_Float16)v;
}

__global__ void k_prep_bias(const float* __restrict__ b_ih,
                            const float* __restrict__ b_hh,
                            float* __restrict__ bsum) {
    int i = blockIdx.x * blockDim.x + threadIdx.x;
    if (i < G4H_) bsum[i] = b_ih[i] + b_hh[i];
}

__global__ void k_prep_d(const float* __restrict__ x, _Float16* __restrict__ d) {
    int idx = blockIdx.x * blockDim.x + threadIdx.x;
    if (idx >= B_ * L_ * C_) return;
    int c  = idx % C_;
    int bp = idx / C_;
    int p  = bp % L_;
    int b  = bp / L_;
    float v = x[((size_t)b * T_ + p + 1) * C_ + c] - x[((size_t)b * T_ + p) * C_ + c];
    d[idx] = (_Float16)v;
}

__global__ void k_init_state(_Float16* __restrict__ h0, unsigned int* __restrict__ sync) {
    int i = blockIdx.x * blockDim.x + threadIdx.x;
    if (i < NSEQ_ * H_) h0[i] = (_Float16)0.0f;
    if (i == 0) *sync = 0u;
}

// ---------------- persistent fused LSTM ----------------
// grid (4,16), block 128 (4 waves). Wave owns 32 sequence rows x 16 h-cols,
// all 4 gates. Weights for the block's 64 gate-columns live in LDS for the
// whole recurrence; c-state and the time-pool sum live in registers.
// Steps are separated by a device-wide atomic barrier (64 co-resident blocks).
__global__ void __launch_bounds__(128)
k_lstm_persistent(const _Float16* __restrict__ d,
                  const _Float16* __restrict__ Wc,
                  const float* __restrict__ bsum,
                  _Float16* __restrict__ hbuf0,
                  _Float16* __restrict__ hbuf1,
                  float* __restrict__ hsum,
                  unsigned int* __restrict__ sync) {
    __shared__ _Float16 Bsh[64 * BPAD_];   // 64 gate-cols x K=384 (+pad) ~49KB

    const int tid  = threadIdx.x;
    const int lane = tid & 31;
    const int wid  = tid >> 5;
    const int j0   = blockIdx.y * 16;      // h-column tile of this block

    // ---- stage step-invariant weight tile into LDS (once) ----
    for (int i = tid; i < 64 * (KTOT_ / 8); i += 128) {
        int c  = i / (KTOT_ / 8);          // local col 0..63 (gate*16 + jcol)
        int kk = (i % (KTOT_ / 8)) * 8;
        int gc = (c >> 4) * H_ + j0 + (c & 15);
        *(half8*)(Bsh + c * BPAD_ + kk) = *(const half8*)(Wc + (size_t)gc * KTOT_ + kk);
    }
    __syncthreads();

    const int rowbase = blockIdx.x * 128 + wid * 32;    // wave's 32 rows
    const int w       = rowbase >> 8;                   // window 0/1 (uniform)
    const int wStart  = w * STRIDE_;
    const int wEnd    = wStart + WIN_;
    const int kb      = (lane >> 4) * 8;
    const int j       = j0 + (lane & 15);
    const int mbase   = (lane >> 4) * 8;

    const float bi = bsum[0 * H_ + j];
    const float bf = bsum[1 * H_ + j];
    const float bg = bsum[2 * H_ + j];
    const float bo = bsum[3 * H_ + j];

    const int n0 = rowbase + (lane & 15);               // row group 0 lane row
    const int n1 = n0 + 16;                             // row group 1 lane row
    const _Float16* dbase0 = d + (size_t)(n0 & (B_ - 1)) * L_ * C_ + kb;
    const _Float16* dbase1 = d + (size_t)(n1 & (B_ - 1)) * L_ * C_ + kb;
    const _Float16* Bsh_l  = Bsh + (size_t)(lane & 15) * BPAD_ + kb;

    float cstr[2][8], hsr[2][8];
#pragma unroll
    for (int rg = 0; rg < 2; ++rg)
#pragma unroll
        for (int r = 0; r < 8; ++r) { cstr[rg][r] = 0.0f; hsr[rg][r] = 0.0f; }

    _Float16* hprev = hbuf0;
    _Float16* hnext = hbuf1;

    for (int t = 0; t < L_; ++t) {
        float8 acc[2][4];
#pragma unroll
        for (int rg = 0; rg < 2; ++rg)
#pragma unroll
            for (int ga = 0; ga < 4; ++ga) acc[rg][ga] = (float8){};

        // --- input-projection K-tiles (input is 0 outside window: skip) ---
        if (t >= wStart && t < wEnd) {
            const _Float16* a0 = dbase0 + (size_t)t * C_;
            const _Float16* a1 = dbase1 + (size_t)t * C_;
#pragma unroll
            for (int kt = 0; kt < 4; ++kt) {
                half16 A0 = load_frag16(a0 + kt * 32);
                half16 A1 = load_frag16(a1 + kt * 32);
#pragma unroll
                for (int ga = 0; ga < 4; ++ga) {
                    half16 Bf = load_frag16(Bsh_l + (size_t)ga * (16 * BPAD_) + kt * 32);
                    acc[0][ga] = WMMA_F16(A0, Bf, acc[0][ga]);
                    acc[1][ga] = WMMA_F16(A1, Bf, acc[1][ga]);
                }
            }
        }
        // --- recurrent K-tiles (always) ---
        {
            const _Float16* h0p = hprev + (size_t)n0 * H_ + kb;
            const _Float16* h1p = hprev + (size_t)n1 * H_ + kb;
#pragma unroll
            for (int kt = 0; kt < 8; ++kt) {
                half16 A0 = load_frag16(h0p + kt * 32);
                half16 A1 = load_frag16(h1p + kt * 32);
#pragma unroll
                for (int ga = 0; ga < 4; ++ga) {
                    half16 Bf = load_frag16(Bsh_l + (size_t)ga * (16 * BPAD_) + C_ + kt * 32);
                    acc[0][ga] = WMMA_F16(A0, Bf, acc[0][ga]);
                    acc[1][ga] = WMMA_F16(A1, Bf, acc[1][ga]);
                }
            }
        }

        // --- gates + cell update in registers; only h goes to memory ---
#pragma unroll
        for (int rg = 0; rg < 2; ++rg) {
#pragma unroll
            for (int r = 0; r < 8; ++r) {
                const int n = rowbase + rg * 16 + mbase + r;
                float iv = sigmoidf_(acc[rg][0][r] + bi);
                float fv = sigmoidf_(acc[rg][1][r] + bf);
                float gv = tanhf(acc[rg][2][r] + bg);
                float ov = sigmoidf_(acc[rg][3][r] + bo);
                float cn = fv * cstr[rg][r] + iv * gv;
                cstr[rg][r] = cn;
                float hn = ov * tanhf(cn);
                hnext[(size_t)n * H_ + j] = (_Float16)hn;
                hsr[rg][r] += hn;
            }
        }

        // --- device-wide step barrier (64 co-resident blocks) ---
        __threadfence();
        __syncthreads();
        if (tid == 0) {
            __hip_atomic_fetch_add(sync, 1u, __ATOMIC_ACQ_REL, __HIP_MEMORY_SCOPE_AGENT);
            const unsigned target = (unsigned)NBLK_ * (unsigned)(t + 1);
            while (__hip_atomic_load(sync, __ATOMIC_ACQUIRE, __HIP_MEMORY_SCOPE_AGENT) < target) {
                __builtin_amdgcn_s_sleep(2);
            }
        }
        __syncthreads();

        _Float16* tmp = hprev; hprev = hnext; hnext = tmp;
    }

    // --- write pooled sums ---
#pragma unroll
    for (int rg = 0; rg < 2; ++rg) {
#pragma unroll
        for (int r = 0; r < 8; ++r) {
            const int n = rowbase + rg * 16 + mbase + r;
            hsum[(size_t)n * H_ + j] = hsr[rg][r];
        }
    }
}

// ---------------- final FC (tiny, fp32) ----------------
__global__ void k_fc(const float* __restrict__ hsum,
                     const float* __restrict__ Wfc,
                     const float* __restrict__ bfc,
                     float* __restrict__ out) {
    int b = blockIdx.x;
    int cls = threadIdx.x;
    if (cls >= CLS_) return;
    float acc = bfc[cls];
    for (int k = 0; k < 2 * H_; ++k) {
        int w = k >> 8, jj = k & 255;
        acc += hsum[(size_t)(w * B_ + b) * H_ + jj] * Wfc[(size_t)cls * (2 * H_) + k];
    }
    out[(size_t)b * CLS_ + cls] = acc;
}

extern "C" void kernel_launch(void* const* d_in, const int* in_sizes, int n_in,
                              void* d_out, int out_size, void* d_ws, size_t ws_size,
                              hipStream_t stream) {
    const float* x    = (const float*)d_in[0];
    const float* W_ih = (const float*)d_in[1];
    const float* W_hh = (const float*)d_in[2];
    const float* b_ih = (const float*)d_in[3];
    const float* b_hh = (const float*)d_in[4];
    const float* W_fc = (const float*)d_in[5];
    const float* b_fc = (const float*)d_in[6];
    float* out = (float*)d_out;

    char* ws = (char*)d_ws;
    _Float16*     Wc   = (_Float16*)(ws + 0);
    float*        bsum = (float*)   (ws + 786432);
    _Float16*     dbuf = (_Float16*)(ws + 790528);
    _Float16*     h0   = (_Float16*)(ws + 20385792);
    _Float16*     h1   = (_Float16*)(ws + 20647936);
    float*        hsum = (float*)   (ws + 20910080);
    unsigned int* sync = (unsigned int*)(ws + 21434368);

    k_prep_w<<<(G4H_ * KTOT_ + 255) / 256, 256, 0, stream>>>(W_ih, W_hh, Wc);
    k_prep_bias<<<4, 256, 0, stream>>>(b_ih, b_hh, bsum);
    k_prep_d<<<(B_ * L_ * C_ + 255) / 256, 256, 0, stream>>>(x, dbuf);
    k_init_state<<<(NSEQ_ * H_ + 255) / 256, 256, 0, stream>>>(h0, sync);

    k_lstm_persistent<<<dim3(4, 16), 128, 0, stream>>>(dbuf, Wc, bsum, h0, h1, hsum, sync);

    k_fc<<<B_, 64, 0, stream>>>(hsum, W_fc, b_fc, out);
}
